// SwinVisionEncoder_82016695484775
// MI455X (gfx1250) — matrix operand order repaired and checked by reading, not compiled, not run on hardware
//
#include <hip/hip_runtime.h>
#include <hip/hip_bf16.h>

// ============================================================================
// Swin-T forward for MI455X (gfx1250, wave32, WMMA).
// Bandwidth-bound workload (~1.5-2 GB activation traffic vs sub-us of matrix
// math): f16 intermediates to halve HBM traffic, f32 accumulation via
// v_wmma_f32_16x16x32_f16 for every GEMM, fused epilogues (bias / exact-GELU /
// residual) to avoid extra passes over activations.
// GEMM: 128x128 block tile, 8 waves, 16KB LDS -> halves A/B panel re-reads vs
// a 64x64 tile; global_prefetch_b8 of the next K-step overlaps loads w/ WMMA.
// Attention: one wave per (window, head); 49x32 tiles padded to 64, QK^T and
// P*V both on WMMA, softmax in-register using 16-lane shfl_xor reductions
// (matches the C/D fragment's 16-lane row groups on wave32).
// ============================================================================

typedef __attribute__((ext_vector_type(16))) _Float16 v16h;
typedef __attribute__((ext_vector_type(8)))  float    v8f;

#define WMMA16x16x32(A, B, Cacc) \
  __builtin_amdgcn_wmma_f32_16x16x32_f16(false, (A), false, (B), (short)0, (Cacc), false, false)

// ---------------------------------------------------------------------------
// Patch embed: 4x4 stride-4 conv as a 48-MAC gather per output element.
// ---------------------------------------------------------------------------
__global__ __launch_bounds__(256) void patch_kernel(
    const float* __restrict__ img, const float* __restrict__ w,
    const float* __restrict__ pb, float* __restrict__ outp)
{
  long i = (long)blockIdx.x * blockDim.x + threadIdx.x;   // (b,y,x,c)
  int c = (int)(i % 96); long t = i / 96;
  int x = (int)(t % 56); t /= 56;
  int y = (int)(t % 56); long b = t / 56;
  if (b >= 32) return;
  float s = pb[c];
  const float* wb = w + (long)c * 48;                      // w[c][ci][ky][kx]
  #pragma unroll
  for (int ci = 0; ci < 3; ++ci)
    #pragma unroll
    for (int ky = 0; ky < 4; ++ky)
      #pragma unroll
      for (int kx = 0; kx < 4; ++kx)
        s += img[((b * 3 + ci) * 224 + (4 * y + ky)) * 224 + (4 * x + kx)]
           * wb[ci * 16 + ky * 4 + kx];
  outp[i] = s;
}

// ---------------------------------------------------------------------------
// LayerNorm: one wave per row, shfl_xor tree reduction (warpSize == 32).
// ---------------------------------------------------------------------------
template <typename TO>
__global__ __launch_bounds__(32) void ln_kernel(
    const float* __restrict__ x, const float* __restrict__ g,
    const float* __restrict__ bt, TO* __restrict__ y, int C)
{
  long row = blockIdx.x;
  const float* xr = x + row * (long)C;
  int lane = threadIdx.x;
  float s = 0.f, ss = 0.f;
  for (int i = lane; i < C; i += 32) { float v = xr[i]; s += v; ss += v * v; }
  #pragma unroll
  for (int off = 16; off; off >>= 1) {
    s  += __shfl_xor(s,  off, 32);
    ss += __shfl_xor(ss, off, 32);
  }
  float mean = s / C;
  float var  = ss / C - mean * mean;
  float inv  = rsqrtf(var + 1e-5f);
  TO* yr = y + row * (long)C;
  for (int i = lane; i < C; i += 32)
    yr[i] = (TO)((xr[i] - mean) * inv * g[i] + bt[i]);
}

// ---------------------------------------------------------------------------
// Generic WMMA GEMM:  Out[M,N] = act(A[M,K] @ W[N,K]^T + bias) (+ resid)
// Block = 256 threads (8 waves), tile 128x128, K-step 32. A/W converted to
// f16 in LDS (16KB). ACT: 0=none, 1=exact GELU. Next K-step prefetched.
// ---------------------------------------------------------------------------
template <typename TA, typename TO, int ACT, bool HAS_BIAS, bool HAS_RES>
__global__ __launch_bounds__(256) void gemm_kernel(
    const TA* __restrict__ A, const float* __restrict__ Wt,
    const float* __restrict__ bias, const float* __restrict__ resid,
    TO* __restrict__ Out, int M, int N, int K)
{
  __shared__ _Float16 As[128][32];
  __shared__ _Float16 Bs[128][32];  // stored as W rows: Bs[n][k] == B^T[n][k]

  const int tid  = threadIdx.x;
  const int wave = tid >> 5;            // 0..7 -> 16-row band
  const int lane = tid & 31;
  const int lg   = lane >> 4;
  const int ll   = lane & 15;
  const long m0  = (long)blockIdx.y * 128;
  const long n0  = (long)blockIdx.x * 128;

  const int arow = tid >> 1;            // 0..127
  const int acol = (tid & 1) * 16;      // 0 or 16

  v8f acc[8];
  #pragma unroll
  for (int nt = 0; nt < 8; ++nt)
    #pragma unroll
    for (int r = 0; r < 8; ++r) acc[nt][r] = 0.f;

  const long gm = m0 + arow;
  const long gn = n0 + arow;

  for (int kt = 0; kt < K; kt += 32) {
    __syncthreads();
    if (gm < M) {
      const TA* src = A + gm * (long)K + kt + acol;
      #pragma unroll
      for (int i = 0; i < 16; ++i) As[arow][acol + i] = (_Float16)(float)src[i];
    } else {
      #pragma unroll
      for (int i = 0; i < 16; ++i) As[arow][acol + i] = (_Float16)0.f;
    }
    if (gn < N) {
      const float* src = Wt + gn * (long)K + kt + acol;
      #pragma unroll
      for (int i = 0; i < 16; ++i) Bs[arow][acol + i] = (_Float16)src[i];
    } else {
      #pragma unroll
      for (int i = 0; i < 16; ++i) Bs[arow][acol + i] = (_Float16)0.f;
    }
    __syncthreads();

    // Prefetch next K-step while this step's WMMAs execute.
    if (kt + 32 < K) {
      if (gm < M) __builtin_prefetch(A  + gm * (long)K + kt + 32 + acol, 0, 1);
      if (gn < N) __builtin_prefetch(Wt + gn * (long)K + kt + 32 + acol, 0, 1);
    }

    // A fragment: rows [16*wave, 16*wave+16); lane-half holds K chunks
    v16h a;
    const int m = 16 * wave + ll;
    #pragma unroll
    for (int e = 0; e < 8; ++e) {
      a[e]     = As[m][8 * lg + e];
      a[8 + e] = As[m][16 + 8 * lg + e];
    }
    #pragma unroll
    for (int nt = 0; nt < 8; ++nt) {
      v16h bb;
      const int n = 16 * nt + ll;
      #pragma unroll
      for (int e = 0; e < 16; ++e) bb[e] = Bs[n][(e < 8 ? 8 * lg + e : 16 + 8 * lg + (e - 8))];
      acc[nt] = WMMA16x16x32(a, bb, acc[nt]);
    }
  }

  // Epilogue: C/D layout -> row = m0 + 16*wave + r + 8*lg, col = n0+16*nt+ll
  #pragma unroll
  for (int nt = 0; nt < 8; ++nt) {
    long col = n0 + 16 * nt + ll;
    if (col >= N) continue;
    float bv = HAS_BIAS ? bias[col] : 0.f;
    #pragma unroll
    for (int r = 0; r < 8; ++r) {
      long row = m0 + 16 * wave + r + 8 * lg;
      if (row >= M) continue;
      float v = acc[nt][r] + bv;
      if (ACT == 1) v = 0.5f * v * (1.0f + erff(v * 0.70710678118654752f));
      if (HAS_RES) v += resid[row * (long)N + col];
      Out[row * (long)N + col] = (TO)v;
    }
  }
}

// ---------------------------------------------------------------------------
// Window attention: 1 wave per (window, head). hd == 32 in all stages.
// Shift/roll handled via index math on the gather (and identical scatter).
// ---------------------------------------------------------------------------
__global__ __launch_bounds__(32) void attn_kernel(
    const _Float16* __restrict__ qkv, const float* __restrict__ rpb,
    _Float16* __restrict__ out, int H, int W, int C, int heads, int shift)
{
  __shared__ _Float16 Qs[64][32];
  __shared__ _Float16 Ks[64][32];
  __shared__ _Float16 Vs[64][32];
  __shared__ _Float16 Ps[64][64];
  __shared__ float    rpbs[169];

  const int lane = threadIdx.x;
  const int lg = lane >> 4, ll = lane & 15;
  const int nW = W / 7;
  const int nWin = (H / 7) * nW;
  const int win = blockIdx.x;
  const int hh  = blockIdx.y;
  const int b = win / nWin;
  const int wrem = win - b * nWin;
  const int wy = wrem / nW, wx = wrem - (wrem / nW) * nW;

  for (int i = lane; i < 169; i += 32) rpbs[i] = rpb[i * heads + hh];

  const float qscale = 0.17677669529663689f;  // 32^-0.5
  const int C3 = 3 * C;
  #pragma unroll
  for (int rr = 0; rr < 2; ++rr) {
    int n = lane + 32 * rr;                    // padded row 0..63
    if (n < 49) {
      int iy = n / 7, ix = n - (n / 7) * 7;
      int hs  = (wy * 7 + iy + shift) % H;
      int wsr = (wx * 7 + ix + shift) % W;
      long row = ((long)b * H + hs) * W + wsr;
      const _Float16* base = qkv + row * (long)C3 + hh * 32;
      #pragma unroll
      for (int d = 0; d < 32; ++d) {
        Qs[n][d] = (_Float16)((float)base[d] * qscale);
        Ks[n][d] = base[C + d];
        Vs[n][d] = base[2 * C + d];
      }
    } else {
      #pragma unroll
      for (int d = 0; d < 32; ++d) {
        Qs[n][d] = (_Float16)0.f; Ks[n][d] = (_Float16)0.f; Vs[n][d] = (_Float16)0.f;
      }
    }
  }
  __syncthreads();

  // S = Q K^T : 4x4 tiles of 16x16, single K-step of 32
  v8f S[4][4];
  #pragma unroll
  for (int mt = 0; mt < 4; ++mt)
    #pragma unroll
    for (int nt = 0; nt < 4; ++nt)
      #pragma unroll
      for (int r = 0; r < 8; ++r) S[mt][nt][r] = 0.f;

  #pragma unroll
  for (int mt = 0; mt < 4; ++mt) {
    v16h a;
    #pragma unroll
    for (int e = 0; e < 8; ++e) {
      a[e]     = Qs[16 * mt + ll][8 * lg + e];
      a[8 + e] = Qs[16 * mt + ll][16 + 8 * lg + e];
    }
    #pragma unroll
    for (int nt = 0; nt < 4; ++nt) {
      v16h bb;      // B^T[n][k] = K[n][k] -> Ks row-major directly
      #pragma unroll
      for (int e = 0; e < 16; ++e) bb[e] = Ks[16 * nt + ll][(e < 8 ? 8 * lg + e : 16 + 8 * lg + (e - 8))];
      S[mt][nt] = WMMA16x16x32(a, bb, S[mt][nt]);
    }
  }

  // rpb + shift mask + softmax per row (row == 16-lane group), write P
  #pragma unroll
  for (int mt = 0; mt < 4; ++mt) {
    #pragma unroll
    for (int r = 0; r < 8; ++r) {
      int mq = 16 * mt + r + 8 * lg;          // query index (uniform over ll)
      if (mq < 49) {
        int qy = mq / 7, qx = mq - (mq / 7) * 7;
        float vals[4];
        #pragma unroll
        for (int nt = 0; nt < 4; ++nt) {
          int mk = 16 * nt + ll;              // key index
          float s = S[mt][nt][r];
          if (mk < 49) {
            int ky = mk / 7, kx = mk - (mk / 7) * 7;
            s += rpbs[(qy - ky + 6) * 13 + (qx - kx + 6)];
            if (shift > 0) {
              int hq = wy * 7 + qy, wq = wx * 7 + qx;
              int hk = wy * 7 + ky, wk = wx * 7 + kx;
              int rq = (hq < H - 7 ? 0 : (hq < H - 3 ? 1 : 2)) * 3
                     + (wq < W - 7 ? 0 : (wq < W - 3 ? 1 : 2));
              int rk = (hk < H - 7 ? 0 : (hk < H - 3 ? 1 : 2)) * 3
                     + (wk < W - 7 ? 0 : (wk < W - 3 ? 1 : 2));
              if (rq != rk) s -= 100.0f;
            }
          } else {
            s = -1e30f;                       // padded key columns
          }
          vals[nt] = s;
        }
        float mx = fmaxf(fmaxf(vals[0], vals[1]), fmaxf(vals[2], vals[3]));
        #pragma unroll
        for (int off = 1; off < 16; off <<= 1) mx = fmaxf(mx, __shfl_xor(mx, off, 32));
        float sum = 0.f;
        #pragma unroll
        for (int nt = 0; nt < 4; ++nt) { vals[nt] = __expf(vals[nt] - mx); sum += vals[nt]; }
        #pragma unroll
        for (int off = 1; off < 16; off <<= 1) sum += __shfl_xor(sum, off, 32);
        float inv = 1.0f / sum;
        #pragma unroll
        for (int nt = 0; nt < 4; ++nt) Ps[mq][16 * nt + ll] = (_Float16)(vals[nt] * inv);
      } else {
        #pragma unroll
        for (int nt = 0; nt < 4; ++nt) Ps[mq][16 * nt + ll] = (_Float16)0.f;
      }
    }
  }
  __syncthreads();

  // O = P V : 4x2 tiles, K = 64 -> two K-steps of 32
  v8f O[4][2];
  #pragma unroll
  for (int mt = 0; mt < 4; ++mt)
    #pragma unroll
    for (int nt = 0; nt < 2; ++nt)
      #pragma unroll
      for (int r = 0; r < 8; ++r) O[mt][nt][r] = 0.f;

  #pragma unroll
  for (int ks = 0; ks < 2; ++ks) {
    const int kb = 32 * ks;
    #pragma unroll
    for (int mt = 0; mt < 4; ++mt) {
      v16h a;
      #pragma unroll
      for (int e = 0; e < 8; ++e) {
        a[e]     = Ps[16 * mt + ll][kb + 8 * lg + e];
        a[8 + e] = Ps[16 * mt + ll][kb + 16 + 8 * lg + e];
      }
      #pragma unroll
      for (int nt = 0; nt < 2; ++nt) {
        v16h bb;  // B^T[d][m'] = V[m'][d] -> strided read of Vs
        #pragma unroll
        for (int e = 0; e < 16; ++e) bb[e] = Vs[kb + (e < 8 ? 8 * lg + e : 16 + 8 * lg + (e - 8))][16 * nt + ll];
        O[mt][nt] = WMMA16x16x32(a, bb, O[mt][nt]);
      }
    }
  }

  // scatter to (B,H,W,C) with identical roll-back indexing
  #pragma unroll
  for (int mt = 0; mt < 4; ++mt) {
    #pragma unroll
    for (int r = 0; r < 8; ++r) {
      int mq = 16 * mt + r + 8 * lg;
      if (mq < 49) {
        int iy = mq / 7, ix = mq - (mq / 7) * 7;
        int hs  = (wy * 7 + iy + shift) % H;
        int wsr = (wx * 7 + ix + shift) % W;
        long row = ((long)b * H + hs) * W + wsr;
        #pragma unroll
        for (int nt = 0; nt < 2; ++nt)
          out[row * (long)C + hh * 32 + 16 * nt + ll] = (_Float16)O[mt][nt][r];
      }
    }
  }
}

// ---------------------------------------------------------------------------
// Patch merge: gather 2x2 concat (x0,x1,x2,x3 order) + LN -> f16 rows of 4C.
// ---------------------------------------------------------------------------
__global__ __launch_bounds__(32) void merge_ln_kernel(
    const float* __restrict__ X, const float* __restrict__ g,
    const float* __restrict__ bt, _Float16* __restrict__ y,
    int H, int W, int C)
{
  const int H2 = H / 2, W2 = W / 2, C4 = 4 * C;
  long row = blockIdx.x;
  int b = (int)(row / ((long)H2 * W2));
  int rem = (int)(row - (long)b * H2 * W2);
  int yy = rem / W2, xx = rem - (rem / W2) * W2;
  int lane = threadIdx.x;

  float s = 0.f, ss = 0.f;
  for (int j = lane; j < C4; j += 32) {
    int q = j / C, cj = j - q * C;
    int dy = q & 1, dx = q >> 1;             // q: 0=(0,0) 1=(1,0) 2=(0,1) 3=(1,1)
    float v = X[(((long)b * H + 2 * yy + dy) * W + (2 * xx + dx)) * C + cj];
    s += v; ss += v * v;
  }
  #pragma unroll
  for (int off = 16; off; off >>= 1) {
    s  += __shfl_xor(s,  off, 32);
    ss += __shfl_xor(ss, off, 32);
  }
  float mean = s / C4;
  float var  = ss / C4 - mean * mean;
  float inv  = rsqrtf(var + 1e-5f);
  _Float16* yr = y + row * (long)C4;
  for (int j = lane; j < C4; j += 32) {
    int q = j / C, cj = j - q * C;
    int dy = q & 1, dx = q >> 1;
    float v = X[(((long)b * H + 2 * yy + dy) * W + (2 * xx + dx)) * C + cj];
    yr[j] = (_Float16)((v - mean) * inv * g[j] + bt[j]);
  }
}

// ---------------------------------------------------------------------------
// Global average pool over the final 7x7 grid (49 positions).
// ---------------------------------------------------------------------------
__global__ __launch_bounds__(256) void pool_kernel(
    const _Float16* __restrict__ ln, float* __restrict__ pool, int C)
{
  int i = blockIdx.x * blockDim.x + threadIdx.x;  // b*C + c
  if (i >= 32 * C) return;
  int b = i / C, c = i - b * C;
  float s = 0.f;
  #pragma unroll 7
  for (int p = 0; p < 49; ++p) s += (float)ln[((long)b * 49 + p) * C + c];
  pool[i] = s * (1.0f / 49.0f);
}

// ---------------------------------------------------------------------------
// Host orchestration. Param leaves assumed in setup_inputs() dict insertion
// order: x, patch_w, patch_b, patch_ng, patch_nb, [12 blocks x 13 leaves],
// [3 merges x (ng, nb, w)], norm_g, norm_b, proj_w, proj_b  (174 leaves).
// Workspace layout (needs ~218 MB):
//   [  0MB) X      fp32 features (max 38.6MB)
//   [ 40MB) LN     f16  LN / merge-concat rows (max 19.3MB)
//   [ 60MB) QKV    f16  qkv (max 57.8MB); fp32 patch temp aliases here
//   [120MB) ABUF   f16  attention out (max 19.3MB); fp32 pooled aliases here
//   [140MB) H1     f16  fc1 out (max 77.1MB)
// ---------------------------------------------------------------------------
extern "C" void kernel_launch(void* const* d_in, const int* in_sizes, int n_in,
                              void* d_out, int out_size, void* d_ws, size_t ws_size,
                              hipStream_t stream) {
  (void)in_sizes; (void)n_in; (void)out_size; (void)ws_size;
  const float* x_img = (const float*)d_in[0];
  auto Pp = [&](int i) { return (const float*)d_in[i]; };

  char* ws = (char*)d_ws;
  float*    X     = (float*)   (ws);
  _Float16* LN    = (_Float16*)(ws + (size_t)40  * 1024 * 1024);
  _Float16* QKV   = (_Float16*)(ws + (size_t)60  * 1024 * 1024);
  float*    PATCH = (float*)   (ws + (size_t)60  * 1024 * 1024);  // pre-stage alias
  _Float16* ABUF  = (_Float16*)(ws + (size_t)120 * 1024 * 1024);
  float*    POOL  = (float*)   (ws + (size_t)120 * 1024 * 1024);  // post-stage alias
  _Float16* H1    = (_Float16*)(ws + (size_t)140 * 1024 * 1024);

  const int Bn = 32;

  // Patch embed + LN
  {
    long tot = (long)Bn * 56 * 56 * 96;
    patch_kernel<<<dim3((unsigned)((tot + 255) / 256)), 256, 0, stream>>>(
        x_img, Pp(1), Pp(2), PATCH);
    ln_kernel<float><<<Bn * 56 * 56, 32, 0, stream>>>(PATCH, Pp(3), Pp(4), X, 96);
  }

  int H = 56, W = 56, C = 96;
  const int depths[4]   = {2, 2, 6, 2};
  const int headsArr[4] = {3, 6, 12, 24};
  int blk = 0;

  for (int si = 0; si < 4; ++si) {
    const int heads = headsArr[si];
    for (int bi = 0; bi < depths[si]; ++bi) {
      const int base  = 5 + 13 * blk; ++blk;
      const int shift = (bi % 2 == 1 && H > 7) ? 3 : 0;
      const int M = Bn * H * W;

      // LN1
      ln_kernel<_Float16><<<M, 32, 0, stream>>>(X, Pp(base + 0), Pp(base + 1), LN, C);
      // qkv = LN1 @ qkv_w^T + qkv_b
      {
        int N = 3 * C;
        dim3 g((N + 127) / 128, (M + 127) / 128);
        gemm_kernel<_Float16, _Float16, 0, true, false><<<g, 256, 0, stream>>>(
            LN, Pp(base + 2), Pp(base + 3), nullptr, QKV, M, N, C);
      }
      // windowed attention (roll handled by indexing)
      {
        dim3 g(Bn * (H / 7) * (W / 7), heads);
        attn_kernel<<<g, 32, 0, stream>>>(QKV, Pp(base + 6), ABUF, H, W, C, heads, shift);
      }
      // x = x + attn @ proj_w^T + proj_b
      {
        dim3 g((C + 127) / 128, (M + 127) / 128);
        gemm_kernel<_Float16, float, 0, true, true><<<g, 256, 0, stream>>>(
            ABUF, Pp(base + 4), Pp(base + 5), X, X, M, C, C);
      }
      // LN2
      ln_kernel<_Float16><<<M, 32, 0, stream>>>(X, Pp(base + 7), Pp(base + 8), LN, C);
      // h = GELU(LN2 @ fc1_w^T + fc1_b)
      {
        int N = 4 * C;
        dim3 g((N + 127) / 128, (M + 127) / 128);
        gemm_kernel<_Float16, _Float16, 1, true, false><<<g, 256, 0, stream>>>(
            LN, Pp(base + 9), Pp(base + 10), nullptr, H1, M, N, C);
      }
      // x = x + h @ fc2_w^T + fc2_b
      {
        dim3 g((C + 127) / 128, (M + 127) / 128);
        gemm_kernel<_Float16, float, 0, true, true><<<g, 256, 0, stream>>>(
            H1, Pp(base + 11), Pp(base + 12), X, X, M, C, 4 * C);
      }
    }
    if (si < 3) {
      const int mbase = 5 + 13 * 12 + 3 * si;
      const int H2 = H / 2, W2 = W / 2;
      const int M2 = Bn * H2 * W2;
      merge_ln_kernel<<<M2, 32, 0, stream>>>(X, Pp(mbase + 0), Pp(mbase + 1), LN, H, W, C);
      int N = 2 * C, K = 4 * C;
      dim3 g((N + 127) / 128, (M2 + 127) / 128);
      gemm_kernel<_Float16, float, 0, false, false><<<g, 256, 0, stream>>>(
          LN, Pp(mbase + 2), nullptr, nullptr, X, M2, N, K);
      H = H2; W = W2; C *= 2;
    }
  }

  // Final LN -> global mean pool -> classifier head
  ln_kernel<_Float16><<<Bn * H * W, 32, 0, stream>>>(X, Pp(170), Pp(171), LN, C);
  pool_kernel<<<(Bn * C + 255) / 256, 256, 0, stream>>>(LN, POOL, C);
  {
    dim3 g((256 + 127) / 128, (Bn + 127) / 128);
    gemm_kernel<float, float, 0, true, false><<<g, 256, 0, stream>>>(
        POOL, Pp(172), Pp(173), nullptr, (float*)d_out, Bn, 256, 768);
  }
}
